// WhitenedNoise_25271587570039
// MI455X (gfx1250) — matrix operand order repaired and checked by reading, not compiled, not run on hardware
//
#include <hip/hip_runtime.h>
#include <math.h>

// ---------------------------------------------------------------------------
// WhitenedNoise: out = z @ Lc^T, where Lc = diag(std) @ L is *analytically*
// the Cholesky factor of Sigma = diag(std) (L L^T) diag(std).
//   Lc[i,j] = std[i] * raw[i,j]                      (i > j)
//   Lc[i,i] = std[i] * (softplus(raw[i,i]) + eps)
//   Lc[i,j] = 0                                      (i < j)
// GEMM done with v_wmma_f32_16x16x32_f16 using an f32 -> (f16 hi + f16 lo)
// split (3 WMMAs per K-step: hh + hl + lh) for ~22-bit mantissa accuracy.
// ---------------------------------------------------------------------------

typedef _Float16 v16h __attribute__((ext_vector_type(16)));
typedef _Float16 v8h  __attribute__((ext_vector_type(8)));
typedef _Float16 v4h  __attribute__((ext_vector_type(4)));
typedef float    v8f  __attribute__((ext_vector_type(8)));

#define D_DIM   512
#define EPS_F   1e-6f
#define NWAVES  8            // 256 threads / workgroup
#define ROWP    520          // padded LDS row stride in halves (1040 B -> conflict-free)
#define ZPLANE  (16 * ROWP)  // one 16-row f16 plane
#define WAVE_Z  (2 * ZPLANE) // hi + lo planes per wave
#define BHI_OFF (NWAVES * WAVE_Z)
#define BLO_OFF (BHI_OFF + ZPLANE)
#define LDS_HALVES (BLO_OFF + ZPLANE)   // 149,760 halves = 299,520 B <= 320 KB

union F16x16 { v16h v; v8h h[2]; };

// ---------------------------------------------------------------------------
// Kernel 1: build Lc^T split into f16 hi/lo planes, row-major [n][k], k=0..511.
// BT[n][k] = Lc[n][k]; zero above diagonal so ragged K-tiles read zeros.
// ---------------------------------------------------------------------------
__global__ void __launch_bounds__(256) lc_prep_kernel(
    const float* __restrict__ stdv, const float* __restrict__ raw,
    _Float16* __restrict__ bthi, _Float16* __restrict__ btlo) {
  int i = blockIdx.x * 256 + threadIdx.x;   // 0 .. 512*512-1
  int n = i >> 9;
  int k = i & (D_DIM - 1);
  float v = 0.0f;
  if (k <= n) {
    float r = raw[(size_t)n * D_DIM + k];
    if (k == n) {
      float sp = (r > 20.0f) ? r : log1pf(expf(r));   // stable softplus
      v = (sp + EPS_F) * stdv[n];
    } else {
      v = r * stdv[n];
    }
  }
  _Float16 h = (_Float16)v;
  bthi[i] = h;
  btlo[i] = (_Float16)(v - (float)h);
}

// ---------------------------------------------------------------------------
// Kernel 2: out[m,n] = sum_k z[m,k] * BT[n,k].
// Workgroup = 8 waves; wave w owns rows [m0, m0+16). z strip converted once to
// f16 hi/lo in LDS. Per n-tile, 16 rows of BT (hi/lo) are staged into LDS
// cooperatively, truncated to the triangular K extent; K-tiles above the
// diagonal are skipped entirely.
// ---------------------------------------------------------------------------
__global__ void __launch_bounds__(256, 1) whitened_gemm_kernel(
    const float* __restrict__ z, const _Float16* __restrict__ bthi,
    const _Float16* __restrict__ btlo, float* __restrict__ out) {
  extern __shared__ _Float16 lds[];

  const int tid    = threadIdx.x;
  const int lane   = tid & 31;
  const int waveId = tid >> 5;
  const int m0     = (blockIdx.x * NWAVES + waveId) * 16;

  const int zhiB = waveId * WAVE_Z;
  const int zloB = zhiB + ZPLANE;

  // ---- Phase 1: stage this wave's 16x512 z strip as f16 hi/lo into LDS ----
  {
    const float4* z4 = (const float4*)z;           // 128 float4 per row
    for (int idx = lane; idx < 16 * 128; idx += 32) {
      int r  = idx >> 7;
      int c4 = idx & 127;
      float4 val = z4[(size_t)(m0 + r) * 128 + c4];
      float f[4] = {val.x, val.y, val.z, val.w};
      v4h hi, lo;
#pragma unroll
      for (int e = 0; e < 4; ++e) {
        _Float16 h = (_Float16)f[e];
        hi[e] = h;
        lo[e] = (_Float16)(f[e] - (float)h);
      }
      int off = r * ROWP + c4 * 4;
      *(v4h*)&lds[zhiB + off] = hi;
      *(v4h*)&lds[zloB + off] = lo;
    }
  }

  const int rowA  = lane & 15;                 // M (and B-slab row) index
  const int koffA = (lane >> 4) << 3;          // A frag: lanes16-31 shift K by 8
  const int koffB = (lane >> 4) << 3;          // B frag: lanes16-31 shift K by 8

  // ---- Phase 2: loop over output column tiles (triangular K skipping) ----
  for (int nt = 0; nt < D_DIM / 16; ++nt) {
    const int n0   = nt * 16;
    const int kt32 = (nt >> 1) + 1;            // # of K=32 tiles needed
    const int chunks = kt32 << 2;              // 8-half chunks per BT row

    __syncthreads();                           // z ready / B slab reusable

    // Cooperative load of BT rows n0..n0+15 (hi/lo) into LDS, K-truncated.
    for (int idx = tid; idx < 16 * chunks; idx += 256) {
      int r = idx / chunks;
      int c = idx - r * chunks;
      size_t g = (size_t)(n0 + r) * D_DIM + (size_t)c * 8;
      int off = r * ROWP + c * 8;
      *(v8h*)&lds[BHI_OFF + off] = *(const v8h*)(bthi + g);
      *(v8h*)&lds[BLO_OFF + off] = *(const v8h*)(btlo + g);
    }
    __syncthreads();

    v8f acc = {};
    for (int kt = 0; kt < kt32; ++kt) {
      const int kb = kt << 5;
      const int aoff = rowA * ROWP + kb + koffA;
      const int boff = rowA * ROWP + kb + koffB;

      F16x16 ahi, alo, bhi, blo;
      ahi.h[0] = *(const v8h*)&lds[zhiB + aoff];
      ahi.h[1] = *(const v8h*)&lds[zhiB + aoff + 16];
      alo.h[0] = *(const v8h*)&lds[zloB + aoff];
      alo.h[1] = *(const v8h*)&lds[zloB + aoff + 16];
      bhi.h[0] = *(const v8h*)&lds[BHI_OFF + boff];
      bhi.h[1] = *(const v8h*)&lds[BHI_OFF + boff + 16];
      blo.h[0] = *(const v8h*)&lds[BLO_OFF + boff];
      blo.h[1] = *(const v8h*)&lds[BLO_OFF + boff + 16];

      acc = __builtin_amdgcn_wmma_f32_16x16x32_f16(
          false, ahi.v, false, bhi.v, (short)0, acc, false, false);
      acc = __builtin_amdgcn_wmma_f32_16x16x32_f16(
          false, ahi.v, false, blo.v, (short)0, acc, false, false);
      acc = __builtin_amdgcn_wmma_f32_16x16x32_f16(
          false, alo.v, false, bhi.v, (short)0, acc, false, false);
    }

    // C/D layout: lane holds column n = n0+(lane&15); VGPR v -> row
    // m0 + v + 8*(lane>=16). Each store writes two 64B row segments.
    const int n     = n0 + rowA;
    const int mBase = m0 + ((lane >> 4) << 3);
#pragma unroll
    for (int v = 0; v < 8; ++v) {
      out[(size_t)(mBase + v) * D_DIM + n] = acc[v];
    }
  }
}

// ---------------------------------------------------------------------------
extern "C" void kernel_launch(void* const* d_in, const int* in_sizes, int n_in,
                              void* d_out, int out_size, void* d_ws, size_t ws_size,
                              hipStream_t stream) {
  const float* stdv = (const float*)d_in[0];   // (512,)
  const float* raw  = (const float*)d_in[1];   // (512, 512)
  const float* z    = (const float*)d_in[2];   // (B, 512)
  float* out        = (float*)d_out;           // (B, 512)

  _Float16* bthi = (_Float16*)d_ws;                    // 512 KB
  _Float16* btlo = bthi + (size_t)D_DIM * D_DIM;       // 512 KB

  // Build Lc^T hi/lo planes (1 MB workspace).
  lc_prep_kernel<<<(D_DIM * D_DIM) / 256, 256, 0, stream>>>(stdv, raw, bthi, btlo);

  const int Btot = in_sizes[2] / D_DIM;                // 65536
  const int grid = Btot / (16 * NWAVES);               // 512 workgroups
  const size_t shmem = (size_t)LDS_HALVES * sizeof(_Float16);  // ~293 KB

  whitened_gemm_kernel<<<grid, 256, shmem, stream>>>(z, bthi, btlo, out);
}
